// STGATE_36137854828684
// MI455X (gfx1250) — matrix operand order repaired and checked by reading, not compiled, not run on hardware
//
#include <hip/hip_runtime.h>
#include <hip/hip_bf16.h>
#include <math.h>

typedef __bf16 bf16;
typedef __attribute__((ext_vector_type(16))) __bf16 v16bf;
typedef __attribute__((ext_vector_type(8)))  float  v8f;

#define BATCH 16
#define NDIM  1024
#define CDIM  64
#define NN    (1024LL * 1024LL)

// ---------------------------------------------------------------------------
// CDNA5 async copy: 16 bytes global -> LDS, tracked by ASYNCcnt.
// LDS byte offset = low 32 bits of the generic pointer (flat LDS aperture).
// ---------------------------------------------------------------------------
__device__ __forceinline__ void async_copy_b128(const void* gptr, void* lptr) {
  unsigned loff = (unsigned)(size_t)lptr;
  unsigned long long ga = (unsigned long long)gptr;
  asm volatile("global_load_async_to_lds_b128 %0, %1, off"
               :: "v"(loff), "v"(ga) : "memory");
}
#define S_WAIT_ASYNC(N) asm volatile("s_wait_asynccnt " #N ::: "memory")

// ---------------------------------------------------------------------------
// 1. f_s[b,n] = x[b,n,:]·W1 ;  f_t[b,n] = x[b,n,:]·W3   (one pass over x)
// ---------------------------------------------------------------------------
__global__ __launch_bounds__(256) void stgate_proj(
    const float* __restrict__ x, const float* __restrict__ W1,
    const float* __restrict__ W3, float* __restrict__ fs, float* __restrict__ ft)
{
  __shared__ float r1[256], r2[256];
  int row = blockIdx.x, tid = threadIdx.x;
  const float* xr = x + (size_t)row * NDIM;
  float a = 0.f, b = 0.f;
#pragma unroll
  for (int p = 0; p < 4; ++p) {
    int i = tid + p * 256;
    float v = xr[i];
    a += v * W1[i];
    b += v * W3[i];
  }
  r1[tid] = a; r2[tid] = b; __syncthreads();
  for (int s = 128; s > 0; s >>= 1) {
    if (tid < s) { r1[tid] += r1[tid + s]; r2[tid] += r2[tid + s]; }
    __syncthreads();
  }
  if (tid == 0) { fs[row] = r1[0]; ft[row] = r2[0]; }
}

// ---------------------------------------------------------------------------
// 2. Mt[b,k,j] = bf16( tanh(f[b,j]*Wrow[k] + bias[j,k]) )   (TRANSPOSED out)
//    64x64 tile transpose through LDS: coalesced bias reads + Mt writes.
// ---------------------------------------------------------------------------
__global__ __launch_bounds__(256) void stgate_build_mt(
    const float* __restrict__ f, const float* __restrict__ Wrow,
    const float* __restrict__ bias, bf16* __restrict__ Mt)
{
  __shared__ bf16 tile[64][66];
  int k0 = blockIdx.x * 64;
  int j0 = blockIdx.y * 64;
  int b  = blockIdx.z;
  int tid = threadIdx.x;
#pragma unroll
  for (int q = 0; q < 16; ++q) {
    int e = q * 256 + tid;
    int jj = e >> 6, kk = e & 63;
    float v = tanhf(f[b * NDIM + j0 + jj] * Wrow[k0 + kk] +
                    bias[(size_t)(j0 + jj) * NDIM + k0 + kk]);
    tile[jj][kk] = (bf16)v;
  }
  __syncthreads();
#pragma unroll
  for (int q = 0; q < 16; ++q) {
    int e = q * 256 + tid;
    int kk = e >> 6, jj = e & 63;
    Mt[(size_t)b * NN + (size_t)(k0 + kk) * NDIM + (j0 + jj)] = tile[jj][kk];
  }
}

// f32 -> bf16 conversion (row-major copy)
__global__ __launch_bounds__(256) void stgate_cvt_bf16(
    const float* __restrict__ s, bf16* __restrict__ d, long long n)
{
  long long i = (long long)blockIdx.x * 256 + threadIdx.x;
  if (i < n) d[i] = (bf16)s[i];
}

// gat_W [1024][64] -> bf16 transpose [64][1024]
__global__ __launch_bounds__(256) void stgate_cvt_t64(
    const float* __restrict__ src, bf16* __restrict__ dst)
{
  int idx = blockIdx.x * 256 + threadIdx.x;   // < 65536
  int t = idx >> 6;
  int c = idx & 63;
  dst[(size_t)c * NDIM + t] = (bf16)src[idx];
}

// ---------------------------------------------------------------------------
// 3. Batched bf16 GEMM: C[bz] = A[bz] (MxK, row-major) @ BT[bz]^T
//    (BT is Nout x K row-major). 128x64 tile, BK=64, double-buffered LDS
//    filled with global_load_async_to_lds_b128; 8 WMMAs / wave / stage.
// ---------------------------------------------------------------------------
#define BM 128
#define BN 64
#define BKK 64
#define LPAD 8

__global__ __launch_bounds__(256) void stgate_gemm_bf16(
    const bf16* __restrict__ A, long long sA,
    const bf16* __restrict__ BT, long long sB,
    float* __restrict__ Cmat, long long sC,
    int Nout, int K)
{
  __shared__ bf16 lA[2][BM][BKK + LPAD];   // 2*128*72*2 = 36 KB
  __shared__ bf16 lB[2][BN][BKK + LPAD];   // 2* 64*72*2 = 18 KB
  int bz = blockIdx.z;
  const bf16* Ab  = A  + (long long)bz * sA;
  const bf16* BTb = BT + (long long)bz * sB;
  float* Cb = Cmat + (long long)bz * sC;
  int m0 = blockIdx.y * BM;
  int n0 = blockIdx.x * BN;
  int tid  = threadIdx.x;
  int lane = tid & 31;
  int wid  = tid >> 5;
  int wm   = wid >> 1;      // 0..3 -> 32-row group
  int wn   = wid & 1;       // 0..1 -> 32-col group
  int half = lane >> 4;
  int l16  = lane & 15;

  v8f acc[2][2] = {};

  auto stage = [&](int buf, int k0) {
    // A tile: 128x64 bf16 = 1024 16B-chunks, 4 per thread
#pragma unroll
    for (int q = 0; q < 4; ++q) {
      int chunk = q * 256 + tid;
      int r = chunk >> 3;
      int c = (chunk & 7) << 3;
      async_copy_b128(Ab + (size_t)(m0 + r) * K + (k0 + c), &lA[buf][r][c]);
    }
    // B tile: 64x64 bf16 = 512 16B-chunks, 2 per thread
#pragma unroll
    for (int q = 0; q < 2; ++q) {
      int chunk = q * 256 + tid;
      int n = chunk >> 3;
      int c = (chunk & 7) << 3;
      async_copy_b128(BTb + (size_t)(n0 + n) * K + (k0 + c), &lB[buf][n][c]);
    }
  };

  const int nIter = K / BKK;
  stage(0, 0);
  for (int it = 0; it < nIter; ++it) {
    if (it + 1 < nIter) {
      stage((it + 1) & 1, (it + 1) * BKK);   // prefetch next stage
      S_WAIT_ASYNC(6);                       // 6 in flight = next stage only
    } else {
      S_WAIT_ASYNC(0);
    }
    __syncthreads();
    int buf = it & 1;

#pragma unroll
    for (int kk = 0; kk < 2; ++kk) {         // two 32-wide K steps
      v16bf af[2], bfrag[2];
#pragma unroll
      for (int i = 0; i < 2; ++i) {
        // A 16x32 layout: elems 0..7 = K[half*8..+7], 8..15 = K[half*8+16..+23]
        const bf16* pa = &lA[buf][wm * 32 + i * 16 + l16][kk * 32 + half * 8];
        v16bf v;
        ((float4*)&v)[0] = *(const float4*)pa;
        ((float4*)&v)[1] = *(const float4*)(pa + 16);
        af[i] = v;
      }
#pragma unroll
      for (int j = 0; j < 2; ++j) {
        // B 32x16 layout: elems 0..7 = K[half*8..+7], 8..15 = K[half*8+16..+23], col l16
        const bf16* pb = &lB[buf][wn * 32 + j * 16 + l16][kk * 32 + half * 8];
        v16bf v;
        ((float4*)&v)[0] = *(const float4*)pb;
        ((float4*)&v)[1] = *(const float4*)(pb + 16);
        bfrag[j] = v;
      }
#pragma unroll
      for (int i = 0; i < 2; ++i)
#pragma unroll
        for (int j = 0; j < 2; ++j)
          acc[i][j] = __builtin_amdgcn_wmma_f32_16x16x32_bf16(
              false, af[i], false, bfrag[j], (short)0, acc[i][j], false, false);
    }
    __syncthreads();
  }

  // store: VGPR r holds row (r + 8*half), col = l16
#pragma unroll
  for (int i = 0; i < 2; ++i)
#pragma unroll
    for (int j = 0; j < 2; ++j) {
      int mb = m0 + wm * 32 + i * 16 + half * 8;
      int nb = n0 + wn * 32 + j * 16 + l16;
#pragma unroll
      for (int r = 0; r < 8; ++r)
        Cb[(size_t)(mb + r) * Nout + nb] = acc[i][j][r];
    }
}

// ---------------------------------------------------------------------------
// 4. Batch-dim standardization in place (mu/var over B=16, ddof=0)
// ---------------------------------------------------------------------------
__global__ __launch_bounds__(256) void stgate_standardize(float* __restrict__ S)
{
  size_t idx = (size_t)blockIdx.x * 256 + threadIdx.x;   // < NN
  float v[BATCH];
  float s = 0.f;
#pragma unroll
  for (int b = 0; b < BATCH; ++b) { v[b] = S[(size_t)b * NN + idx]; s += v[b]; }
  float mu = s * (1.f / BATCH);
  float var = 0.f;
#pragma unroll
  for (int b = 0; b < BATCH; ++b) { float d = v[b] - mu; var += d * d; }
  var *= (1.f / BATCH);
  float r = rsqrtf(var + 1e-5f);
#pragma unroll
  for (int b = 0; b < BATCH; ++b) S[(size_t)b * NN + idx] = (v[b] - mu) * r;
}

// ---------------------------------------------------------------------------
// 5. k-th largest per row via k iterations of max-below-threshold
// ---------------------------------------------------------------------------
__global__ __launch_bounds__(256) void stgate_topk(
    const float* __restrict__ A, const int* __restrict__ pk, float* __restrict__ kth)
{
  __shared__ float vals[1024];
  __shared__ float red[256];
  __shared__ float s_th;
  int row = blockIdx.x, tid = threadIdx.x;
  const float* Arow = A + (size_t)row * NDIM;
#pragma unroll
  for (int p = 0; p < 4; ++p) vals[tid + p * 256] = Arow[tid + p * 256];
  if (tid == 0) s_th = __builtin_inff();
  __syncthreads();
  int k = *pk;
  for (int it = 0; it < k; ++it) {
    float t = s_th;
    float lm = -__builtin_inff();
#pragma unroll
    for (int p = 0; p < 4; ++p) {
      float v = vals[tid + p * 256];
      if (v < t && v > lm) lm = v;
    }
    red[tid] = lm; __syncthreads();
    for (int s = 128; s > 0; s >>= 1) {
      if (tid < s) red[tid] = fmaxf(red[tid], red[tid + s]);
      __syncthreads();
    }
    if (tid == 0) s_th = red[0];
    __syncthreads();
  }
  if (tid == 0) kth[row] = s_th;
}

// 6. Xh = bf16(T_hat * x)   (row-major: A operand of GEMM2)
__global__ __launch_bounds__(256) void stgate_xh(
    const float* __restrict__ T, const float* __restrict__ x, bf16* __restrict__ Xh)
{
  size_t i = (size_t)blockIdx.x * 256 + threadIdx.x;
  Xh[i] = (bf16)(T[i] * x[i]);
}

// 7. h_src/h_dst = H · a_src / a_dst (per row of 64)
__global__ __launch_bounds__(64) void stgate_hdot(
    const float* __restrict__ H, const float* __restrict__ asrc,
    const float* __restrict__ adst, float* __restrict__ hs, float* __restrict__ hd)
{
  __shared__ float r1[64], r2[64];
  int row = blockIdx.x, tid = threadIdx.x;
  float h = H[(size_t)row * CDIM + tid];
  r1[tid] = h * asrc[tid]; r2[tid] = h * adst[tid]; __syncthreads();
  for (int s = 32; s > 0; s >>= 1) {
    if (tid < s) { r1[tid] += r1[tid + s]; r2[tid] += r2[tid + s]; }
    __syncthreads();
  }
  if (tid == 0) { hs[row] = r1[0]; hd[row] = r2[0]; }
}

// ---------------------------------------------------------------------------
// 8. Masked softmax + sparse alpha @ H (only ~top_k j contribute).
//    Deterministic: fixed per-thread j ranges, LDS tree reductions, no atomics.
// ---------------------------------------------------------------------------
__global__ __launch_bounds__(256) void stgate_attn_out(
    const float* __restrict__ A, const float* __restrict__ kth,
    const float* __restrict__ hsrc, const float* __restrict__ hdst,
    const float* __restrict__ H, float* __restrict__ out)
{
  __shared__ float red[256];
  __shared__ float s_max, s_inv;
  int row = blockIdx.x;
  int b = row >> 10;
  int tid = threadIdx.x;
  const float* Arow = A + (size_t)row * NDIM;
  const float* hd = hdst + (size_t)b * NDIM;
  float kv = kth[row];
  float hs = hsrc[row];

  float lmax = -__builtin_inff();
#pragma unroll
  for (int p = 0; p < 4; ++p) {
    int j = tid + p * 256;
    float a = Arow[j];
    if (a >= kv && a > 0.f) {
      float l = hs + hd[j];
      l = l >= 0.f ? l : 0.2f * l;
      lmax = fmaxf(lmax, l);
    }
  }
  red[tid] = lmax; __syncthreads();
  for (int s = 128; s > 0; s >>= 1) {
    if (tid < s) red[tid] = fmaxf(red[tid], red[tid + s]);
    __syncthreads();
  }
  if (tid == 0) s_max = red[0];
  __syncthreads();
  float rmax = s_max;
  float lsum = 0.f;
  if (rmax > -__builtin_inff()) {
#pragma unroll
    for (int p = 0; p < 4; ++p) {
      int j = tid + p * 256;
      float a = Arow[j];
      if (a >= kv && a > 0.f) {
        float l = hs + hd[j];
        l = l >= 0.f ? l : 0.2f * l;
        lsum += __expf(l - rmax);
      }
    }
  }
  __syncthreads();
  red[tid] = lsum; __syncthreads();
  for (int s = 128; s > 0; s >>= 1) {
    if (tid < s) red[tid] += red[tid + s];
    __syncthreads();
  }
  if (tid == 0) s_inv = (red[0] > 0.f) ? 1.f / red[0] : 0.f;
  __syncthreads();
  float inv = s_inv;

  int c  = tid & 63;      // output channel
  int ph = tid >> 6;      // 4 partial sums over j quarters
  float part = 0.f;
  if (rmax > -__builtin_inff()) {
    for (int j = ph * 256; j < ph * 256 + 256; ++j) {
      float a = Arow[j];
      if (a >= kv && a > 0.f) {
        float l = hs + hd[j];
        l = l >= 0.f ? l : 0.2f * l;
        part += __expf(l - rmax) * H[((size_t)(b << 10) + j) * CDIM + c];
      }
    }
  }
  __syncthreads();
  red[tid] = part; __syncthreads();
  if (tid < 64)
    out[(size_t)row * CDIM + tid] =
        (red[tid] + red[tid + 64] + red[tid + 128] + red[tid + 192]) * inv;
}

// 9. classifier: out[b,o] = g[b,:]·fc_W[o,:] + fc_b[o]
__global__ __launch_bounds__(256) void stgate_fc(
    const float* __restrict__ g, const float* __restrict__ W,
    const float* __restrict__ bias, float* __restrict__ out)
{
  __shared__ float red[256];
  int b = blockIdx.x / 3, o = blockIdx.x % 3;
  int tid = threadIdx.x;
  const float* gr = g + (size_t)b * (NDIM * CDIM);
  const float* wr = W + (size_t)o * (NDIM * CDIM);
  float s = 0.f;
  for (int i = tid; i < NDIM * CDIM; i += 256) s += gr[i] * wr[i];
  red[tid] = s; __syncthreads();
  for (int st = 128; st > 0; st >>= 1) {
    if (tid < st) red[tid] += red[tid + st];
    __syncthreads();
  }
  if (tid == 0) out[b * 3 + o] = red[0] + bias[o];
}

// ---------------------------------------------------------------------------
extern "C" void kernel_launch(void* const* d_in, const int* in_sizes, int n_in,
                              void* d_out, int out_size, void* d_ws, size_t ws_size,
                              hipStream_t stream)
{
  const float* x     = (const float*)d_in[0];
  const float* V_s   = (const float*)d_in[1];
  const float* W1    = (const float*)d_in[2];
  const float* W2    = (const float*)d_in[3];
  const float* b_s   = (const float*)d_in[4];
  const float* V_t   = (const float*)d_in[5];
  const float* W3    = (const float*)d_in[6];
  const float* W4    = (const float*)d_in[7];
  const float* b_t   = (const float*)d_in[8];
  const float* gat_W = (const float*)d_in[9];
  const float* a_src = (const float*)d_in[10];
  const float* a_dst = (const float*)d_in[11];
  const float* fc_W  = (const float*)d_in[12];
  const float* fc_b  = (const float*)d_in[13];
  const int*   top_k = (const int*)d_in[14];
  float* out = (float*)d_out;

  char* ws = (char*)d_ws;
  size_t off = 0;
  auto take = [&](size_t bytes) {
    void* p = ws + off;
    off += (bytes + 255) & ~(size_t)255;
    return p;
  };

  float* f_s   = (float*)take((size_t)BATCH * NDIM * sizeof(float));
  float* f_t   = (float*)take((size_t)BATCH * NDIM * sizeof(float));
  float* kth   = (float*)take((size_t)BATCH * NDIM * sizeof(float));
  float* h_src = (float*)take((size_t)BATCH * NDIM * sizeof(float));
  float* h_dst = (float*)take((size_t)BATCH * NDIM * sizeof(float));
  bf16*  V_bf  = (bf16*)take((size_t)NN * sizeof(bf16));
  bf16*  gWT   = (bf16*)take((size_t)NDIM * CDIM * sizeof(bf16)); // transposed
  bf16*  M_bf  = (bf16*)take((size_t)BATCH * NN * sizeof(bf16));  // reused 3x
  float* S_s   = (float*)take((size_t)BATCH * NN * sizeof(float)); // A
  float* S_t   = (float*)take((size_t)BATCH * NN * sizeof(float)); // T_hat
  float* H     = (float*)take((size_t)BATCH * NDIM * CDIM * sizeof(float));
  float* g_out = (float*)take((size_t)BATCH * NDIM * CDIM * sizeof(float));

  dim3 b256(256);

  // projections (single pass over x)
  stgate_proj<<<BATCH * NDIM, b256, 0, stream>>>(x, W1, W3, f_s, f_t);

  // ---- spatial branch: A ----
  stgate_cvt_bf16<<<(unsigned)((NN + 255) / 256), b256, 0, stream>>>(V_s, V_bf, NN);
  stgate_build_mt<<<dim3(16, 16, BATCH), b256, 0, stream>>>(f_s, W2, b_s, M_bf);
  stgate_gemm_bf16<<<dim3(NDIM / BN, NDIM / BM, BATCH), b256, 0, stream>>>(
      V_bf, 0, M_bf, NN, S_s, NN, NDIM, NDIM);
  stgate_standardize<<<(unsigned)(NN / 256), b256, 0, stream>>>(S_s);
  stgate_topk<<<BATCH * NDIM, b256, 0, stream>>>(S_s, top_k, kth);

  // ---- temporal branch: T_hat ----
  stgate_cvt_bf16<<<(unsigned)((NN + 255) / 256), b256, 0, stream>>>(V_t, V_bf, NN);
  stgate_build_mt<<<dim3(16, 16, BATCH), b256, 0, stream>>>(f_t, W4, b_t, M_bf);
  stgate_gemm_bf16<<<dim3(NDIM / BN, NDIM / BM, BATCH), b256, 0, stream>>>(
      V_bf, 0, M_bf, NN, S_t, NN, NDIM, NDIM);
  stgate_standardize<<<(unsigned)(NN / 256), b256, 0, stream>>>(S_t);

  // ---- GAT ----
  stgate_xh<<<(unsigned)(BATCH * NN / 256), b256, 0, stream>>>(S_t, x, M_bf);
  stgate_cvt_t64<<<256, b256, 0, stream>>>(gat_W, gWT);
  stgate_gemm_bf16<<<dim3(CDIM / BN, NDIM / BM, BATCH), b256, 0, stream>>>(
      M_bf, NN, gWT, 0, H, (long long)NDIM * CDIM, CDIM, NDIM);
  stgate_hdot<<<BATCH * NDIM, dim3(64), 0, stream>>>(H, a_src, a_dst, h_src, h_dst);
  stgate_attn_out<<<BATCH * NDIM, b256, 0, stream>>>(S_s, kth, h_src, h_dst, H, g_out);
  stgate_fc<<<BATCH * 3, b256, 0, stream>>>(g_out, fc_W, fc_b, out);

  (void)in_sizes; (void)n_in; (void)out_size; (void)ws_size;
}
